// DeformableFeatureAlignment_44994077392898
// MI455X (gfx1250) — matrix operand order repaired and checked by reading, not compiled, not run on hardware
//
#include <hip/hip_runtime.h>
#include <hip/hip_bf16.h>

// ---------------------------------------------------------------------------
// DeformableFeatureAlignment for MI455X (gfx1250, wave32, WMMA bf16)
// ---------------------------------------------------------------------------
// Shapes: B=4, HF=WF=64, HC=WC=32, CF=CC=F=256, DG=8, KK=9, P=16384 pixels.
// All matmul FLOPs (63 GFLOP total) go through v_wmma_f32_16x16x32_bf16.
// Weights are tiny (<= 2.3 MB) and L2-resident (192 MB L2); activations
// ~76 MB << HBM roofline at 23.3 TB/s, so the GEMMs are matrix-issue bound:
// maximize WMMA per VMEM issue with 32x64 (TM=2 x TN=4) tiles per wave.
// ---------------------------------------------------------------------------

typedef __attribute__((ext_vector_type(16))) __bf16 v16bf;
typedef __attribute__((ext_vector_type(8)))  __bf16 v8bf;
typedef __attribute__((ext_vector_type(8)))  float  v8f;

#define NPIX    16384
#define NPAD_OM 224   // 216 offset/mask channels padded to 14*16

__device__ __forceinline__ float sigmoidf_(float x) {
  return 1.0f / (1.0f + __expf(-x));
}

// WMMA A fragment (16x32 bf16), row-major source, contiguous K order:
//   lanes 0-15 : elems 0..15 = K kb+0..15
//   lanes16-31 : elems 0..15 = K kb+16..31
__device__ __forceinline__ v16bf load_a_frag(const __bf16* __restrict__ arow,
                                             int kb, int half) {
  const __bf16* p0 = arow + kb + half * 16;
  v8bf lo = *(const v8bf*)(p0);
  v8bf hi = *(const v8bf*)(p0 + 8);
  v16bf a;
#pragma unroll
  for (int i = 0; i < 8; ++i) { a[i] = lo[i]; a[8 + i] = hi[i]; }
  return a;
}

__device__ __forceinline__ v16bf zero_a_frag() {
  v16bf a;
#pragma unroll
  for (int i = 0; i < 16; ++i) a[i] = (__bf16)0.0f;
  return a;
}

// ---------------------------------------------------------------------------
// (1a) zero the gap accumulator
// ---------------------------------------------------------------------------
__global__ void dfa_zero(float* __restrict__ p, int n) {
  int i = blockIdx.x * 256 + threadIdx.x;
  if (i < n) p[i] = 0.0f;
}

// ---------------------------------------------------------------------------
// (1b) partial GAP sums: grid (B, 16) x 256 threads; each block reduces a
// 256-pixel segment for all 256 channels, atomically accumulated into gap.
// ---------------------------------------------------------------------------
__global__ void dfa_gap_partial(const float* __restrict__ fine,
                                float* __restrict__ gap) {
  const int b = blockIdx.x;
  const int seg = blockIdx.y;
  const int c = threadIdx.x;
  const float* fb = fine + ((size_t)b * 4096 + (size_t)seg * 256) * 256;
  float s = 0.0f;
  for (int p = 0; p < 256; ++p) s += fb[(size_t)p * 256 + c];
  atomicAdd(&gap[b * 256 + c], s);
}

// ---------------------------------------------------------------------------
// (1c) attention: scale[b][c] = 1 + sigmoid(mean_gap @ fs_attn_w)
// ---------------------------------------------------------------------------
__global__ void dfa_attn(const float* __restrict__ gap,
                         const float* __restrict__ fs_attn_w,
                         float* __restrict__ scale) {
  __shared__ float g[256];
  const int b = blockIdx.x;
  const int c = threadIdx.x;
  g[c] = gap[b * 256 + c] * (1.0f / 4096.0f);
  __syncthreads();
  float a = 0.0f;
  for (int k = 0; k < 256; ++k) a += g[k] * fs_attn_w[(size_t)k * 256 + c];
  scale[b * 256 + c] = 1.0f + sigmoidf_(a);
}

// ---------------------------------------------------------------------------
// (2a) A1[p][c] = bf16(fine[p][c] * scale[b][c])
// ---------------------------------------------------------------------------
__global__ void dfa_prep_a1(const float* __restrict__ fine,
                            const float* __restrict__ scale,
                            __bf16* __restrict__ A1) {
  size_t i = (size_t)blockIdx.x * 256 + threadIdx.x;   // over 16384*256
  int c = (int)(i & 255);
  int b = (int)(i >> 20);
  A1[i] = (__bf16)(fine[i] * scale[b * 256 + c]);
}

// ---------------------------------------------------------------------------
// Weight transpose: Bt[n][k] = bf16(W[k][n]), n in [0,Npad) zero-padded.
// ---------------------------------------------------------------------------
__global__ void dfa_transpose_w(const float* __restrict__ W, __bf16* __restrict__ Bt,
                                int K, int N, int Npad) {
  size_t i = (size_t)blockIdx.x * 256 + threadIdx.x;
  if (i >= (size_t)Npad * K) return;
  int n = (int)(i / K);
  int k = (int)(i - (size_t)n * K);
  Bt[i] = (n < N) ? (__bf16)W[(size_t)k * N + n] : (__bf16)0.0f;
}

// ---------------------------------------------------------------------------
// (3a) Half-pixel bilinear upsample 32x32 -> 64x64 (edge clamp).
// Writes cup f32 and fc[:,256+c] = bf16(2*cup).  grid: 16384 x 256
// ---------------------------------------------------------------------------
__global__ void dfa_upsample(const float* __restrict__ coarse,
                             float* __restrict__ cup,
                             __bf16* __restrict__ fc) {
  const int p = blockIdx.x;
  const int c = threadIdx.x;
  const int x = p & 63, y = (p >> 6) & 63, b = p >> 12;
  float sy = 0.5f * (float)y - 0.25f;
  float sx = 0.5f * (float)x - 0.25f;
  float fy = floorf(sy), fx = floorf(sx);
  float wy = sy - fy, wx = sx - fx;
  int y0 = (int)fy, x0 = (int)fx;
  int y0c = y0 < 0 ? 0 : (y0 > 31 ? 31 : y0);
  int y1c = (y0 + 1) < 0 ? 0 : ((y0 + 1) > 31 ? 31 : (y0 + 1));
  int x0c = x0 < 0 ? 0 : (x0 > 31 ? 31 : x0);
  int x1c = (x0 + 1) < 0 ? 0 : ((x0 + 1) > 31 ? 31 : (x0 + 1));
  const float* cb = coarse + (size_t)b * 32 * 32 * 256 + c;
  float v00 = cb[(size_t)((y0c << 5) + x0c) * 256];
  float v01 = cb[(size_t)((y0c << 5) + x1c) * 256];
  float v10 = cb[(size_t)((y1c << 5) + x0c) * 256];
  float v11 = cb[(size_t)((y1c << 5) + x1c) * 256];
  float v = v00 * (1.f - wy) * (1.f - wx) + v01 * (1.f - wy) * wx +
            v10 * wy * (1.f - wx) + v11 * wy * wx;
  cup[(size_t)p * 256 + c] = v;
  fc[(size_t)p * 512 + 256 + c] = (__bf16)(2.0f * v);
}

// ---------------------------------------------------------------------------
// f32 -> bf16 copy with destination stride (fine_cal -> fc[:,:256])
// ---------------------------------------------------------------------------
__global__ void dfa_cvt_strided(const float* __restrict__ src, __bf16* __restrict__ dst,
                                int srcCols, int dstStride) {
  size_t i = (size_t)blockIdx.x * 256 + threadIdx.x;
  int row = (int)(i / srcCols);
  int col = (int)(i - (size_t)row * srcCols);
  dst[(size_t)row * dstStride + col] = (__bf16)src[i];
}

// ---------------------------------------------------------------------------
// Generic bf16 WMMA GEMM:  C[M,N] = A[M,K] @ Bt[N,K]^T
// One wave per block; wave computes (16*TM) x (16*TN) tile.
// grid = (M/(16*TM), N/(16*TN)), block = 32.
// Per K-step: 2*TM A-chunk loads + TN B loads feed TM*TN WMMAs.
// ---------------------------------------------------------------------------
template <typename OutT, int TM, int TN>
__global__ void dfa_gemm(const __bf16* __restrict__ A, const __bf16* __restrict__ Bt,
                         OutT* __restrict__ C, int M, int N, int K) {
  const int mtile0 = blockIdx.x * TM;
  const int ntile0 = blockIdx.y * TN;
  const int lane = threadIdx.x & 31;
  const int half = lane >> 4;
  const int lm   = lane & 15;

  const __bf16* arow[TM];
#pragma unroll
  for (int m = 0; m < TM; ++m)
    arow[m] = A + (size_t)((mtile0 + m) * 16 + lm) * K;

  v8f acc[TM][TN];
#pragma unroll
  for (int m = 0; m < TM; ++m)
#pragma unroll
    for (int t = 0; t < TN; ++t) acc[m][t] = (v8f){0, 0, 0, 0, 0, 0, 0, 0};

#pragma unroll 2
  for (int kb = 0; kb < K; kb += 32) {
    __builtin_prefetch(arow[0] + kb + 128, 0, 0);   // global_prefetch_b8
    v16bf a[TM];
#pragma unroll
    for (int m = 0; m < TM; ++m) a[m] = load_a_frag(arow[m], kb, half);
#pragma unroll
    for (int t = 0; t < TN; ++t) {
      const __bf16* bptr =
          Bt + (size_t)((ntile0 + t) * 16 + lm) * K + kb + half * 16;
      v16bf b = *(const v16bf*)bptr;
#pragma unroll
      for (int m = 0; m < TM; ++m) {
        acc[m][t] = __builtin_amdgcn_wmma_f32_16x16x32_bf16(
            false, a[m], false, b, (short)0, acc[m][t], false, false);
      }
    }
  }

#pragma unroll
  for (int m = 0; m < TM; ++m) {
    const int mbase = (mtile0 + m) * 16 + half * 8;
#pragma unroll
    for (int t = 0; t < TN; ++t) {
      const int ncol = (ntile0 + t) * 16 + lm;
#pragma unroll
      for (int r = 0; r < 8; ++r) {
        C[(size_t)(mbase + r) * N + ncol] = (OutT)acc[m][t][r];
      }
    }
  }
}

// ---------------------------------------------------------------------------
// (4) Implicit-im2col 3x3 "SAME" conv via WMMA:
//   om[p][n] = sum_{tap,k} align[p+shift(tap)][k] * om_w[tap][k][n] + om_b[n]
// A = align bf16 [16384][512]; Bt = [9][224][512]; C = om f32 [16384][224].
// Wave computes a 32-pixel x 32-channel tile (TM=2, TN=2). A 32-pixel M-strip
// lies inside one image row (W=64), so y-masking is tile-uniform and x edge
// masking is per-lane; WMMA always executes with full EXEC (zero fragments).
// grid (512, 7), block 32.
// ---------------------------------------------------------------------------
__global__ void dfa_conv3x3(const __bf16* __restrict__ A, const __bf16* __restrict__ Bt,
                            const float* __restrict__ om_b, float* __restrict__ C) {
  const int Kc = 512;
  const int mtile0 = blockIdx.x * 2;
  const int ntile0 = blockIdx.y * 2;
  const int lane = threadIdx.x & 31;
  const int half = lane >> 4;
  const int lm   = lane & 15;

  const int mbase = mtile0 * 16;          // 32-aligned pixel index
  const int y = (mbase >> 6) & 63;
  const int b = mbase >> 12;

  v8f acc[2][2];
#pragma unroll
  for (int m = 0; m < 2; ++m)
#pragma unroll
    for (int t = 0; t < 2; ++t) acc[m][t] = (v8f){0, 0, 0, 0, 0, 0, 0, 0};

#pragma unroll
  for (int tap = 0; tap < 9; ++tap) {
    const int dy = tap / 3 - 1;
    const int dx = tap % 3 - 1;
    const int ys = y + dy;
    bool valid[2];
    const __bf16* arow[2];
#pragma unroll
    for (int m = 0; m < 2; ++m) {
      const int xm = (mbase & 63) + m * 16 + lm;   // lane's pixel x
      const int xs = xm + dx;
      valid[m] = ((unsigned)ys < 64u) && ((unsigned)xs < 64u);
      arow[m] = A + (size_t)((b << 12) + (ys << 6) + xs) * Kc;  // deref iff valid
    }
    const __bf16* btap = Bt + (size_t)tap * NPAD_OM * Kc;
    for (int kb = 0; kb < Kc; kb += 32) {
      v16bf a[2];
#pragma unroll
      for (int m = 0; m < 2; ++m)
        a[m] = valid[m] ? load_a_frag(arow[m], kb, half) : zero_a_frag();
#pragma unroll
      for (int t = 0; t < 2; ++t) {
        const __bf16* bptr =
            btap + (size_t)((ntile0 + t) * 16 + lm) * Kc + kb + half * 16;
        v16bf bfrag = *(const v16bf*)bptr;
#pragma unroll
        for (int m = 0; m < 2; ++m) {
          acc[m][t] = __builtin_amdgcn_wmma_f32_16x16x32_bf16(
              false, a[m], false, bfrag, (short)0, acc[m][t], false, false);
        }
      }
    }
  }

#pragma unroll
  for (int m = 0; m < 2; ++m) {
#pragma unroll
    for (int t = 0; t < 2; ++t) {
      const int ncol = (ntile0 + t) * 16 + lm;
      const float bias = (ncol < 216) ? om_b[ncol] : 0.0f;
#pragma unroll
      for (int r = 0; r < 8; ++r) {
        C[(size_t)(mbase + m * 16 + half * 8 + r) * NPAD_OM + ncol] =
            acc[m][t][r] + bias;
      }
    }
  }
}

// ---------------------------------------------------------------------------
// (5) Modulated deformable im2col.
// cols[p][kk*256 + dg*32 + cg] = bilerp(cup group dg at p+tap+offset) * mask
// block (32,8): lane = cg (Cg==32==wave32); grid (16384, 9)
// ---------------------------------------------------------------------------
__global__ void dfa_dcn_cols(const float* __restrict__ om,
                             const float* __restrict__ cup,
                             __bf16* __restrict__ cols) {
  const int p  = blockIdx.x;
  const int u  = blockIdx.y * 8 + threadIdx.y;   // 0..71 = dg*9 + kk
  const int cg = threadIdx.x;                    // 0..31
  const int dg = u / 9;
  const int kk = u - dg * 9;
  const int x = p & 63, y = (p >> 6) & 63, b = p >> 12;

  const float* omp = om + (size_t)p * NPAD_OM;
  const float dy = omp[2 * u];
  const float dx = omp[2 * u + 1];
  const float mk = 2.0f * sigmoidf_(omp[144 + u]);

  const float py = (float)y + (float)(kk / 3 - 1) + dy;
  const float px = (float)x + (float)(kk % 3 - 1) + dx;
  const float y0f = floorf(py), x0f = floorf(px);
  const float wy1 = py - y0f, wx1 = px - x0f;
  const float wy0 = 1.0f - wy1, wx0 = 1.0f - wx1;
  const int y0 = (int)y0f, x0 = (int)x0f;

  const float* base = cup + ((size_t)b << 12) * 256 + dg * 32 + cg;

  auto gather = [&](int yi, int xi) -> float {
    const bool v = ((unsigned)yi < 64u) && ((unsigned)xi < 64u);
    const int yc = yi < 0 ? 0 : (yi > 63 ? 63 : yi);
    const int xc = xi < 0 ? 0 : (xi > 63 ? 63 : xi);
    const float t = base[(size_t)((yc << 6) + xc) * 256];
    return v ? t : 0.0f;
  };

  float val = gather(y0, x0) * wy0 * wx0 + gather(y0, x0 + 1) * wy0 * wx1 +
              gather(y0 + 1, x0) * wy1 * wx0 + gather(y0 + 1, x0 + 1) * wy1 * wx1;

  cols[(size_t)p * 2304 + (size_t)kk * 256 + dg * 32 + cg] = (__bf16)(val * mk);
}

// ---------------------------------------------------------------------------
// (6) out = relu(out_mm + dcn_bias) + fine_cal
// ---------------------------------------------------------------------------
__global__ void dfa_epilogue(const float* __restrict__ out_mm,
                             const float* __restrict__ dcn_bias,
                             const float* __restrict__ fine_cal,
                             float* __restrict__ out) {
  size_t i = (size_t)blockIdx.x * 256 + threadIdx.x;
  int f = (int)(i & 255);
  float v = out_mm[i] + dcn_bias[f];
  out[i] = fmaxf(v, 0.0f) + fine_cal[i];
}

// ---------------------------------------------------------------------------
// Host launcher
// ---------------------------------------------------------------------------
extern "C" void kernel_launch(void* const* d_in, const int* in_sizes, int n_in,
                              void* d_out, int out_size, void* d_ws, size_t ws_size,
                              hipStream_t stream) {
  (void)in_sizes; (void)n_in; (void)out_size; (void)ws_size;
  const float* fine       = (const float*)d_in[0];   // [4,64,64,256]
  const float* coarse     = (const float*)d_in[1];   // [4,32,32,256]
  const float* fs_attn_w  = (const float*)d_in[2];   // [256,256]
  const float* fs_conv_w  = (const float*)d_in[3];   // [256,256]
  const float* offset_w   = (const float*)d_in[4];   // [512,512]
  const float* om_w       = (const float*)d_in[5];   // [3,3,512,216]
  const float* om_b       = (const float*)d_in[6];   // [216]
  const float* dcn_kernel = (const float*)d_in[7];   // [2304,256]
  const float* dcn_bias   = (const float*)d_in[8];   // [256]
  float* out = (float*)d_out;                        // [4,64,64,256]

  // --- workspace carve-out (256B aligned segments) ---
  char* base = (char*)d_ws;
  size_t off = 0;
  auto carve = [&](size_t bytes) -> char* {
    char* p = base + off;
    off += (bytes + 255) & ~(size_t)255;
    return p;
  };
  float*  gap        = (float*)  carve(4 * 256 * sizeof(float));
  float*  scale      = (float*)  carve(4 * 256 * sizeof(float));
  __bf16* A1         = (__bf16*) carve((size_t)NPIX * 256 * 2);
  __bf16* Bt_fsconv  = (__bf16*) carve((size_t)256 * 256 * 2);
  float*  fine_cal   = (float*)  carve((size_t)NPIX * 256 * 4);
  float*  cup        = (float*)  carve((size_t)NPIX * 256 * 4);
  __bf16* fc         = (__bf16*) carve((size_t)NPIX * 512 * 2);
  __bf16* Bt_off     = (__bf16*) carve((size_t)512 * 512 * 2);
  __bf16* align_bf   = (__bf16*) carve((size_t)NPIX * 512 * 2);
  __bf16* Bt_om      = (__bf16*) carve((size_t)9 * NPAD_OM * 512 * 2);
  float*  om         = (float*)  carve((size_t)NPIX * NPAD_OM * 4);
  __bf16* cols       = (__bf16*) carve((size_t)NPIX * 2304 * 2);
  __bf16* Bt_dcn     = (__bf16*) carve((size_t)256 * 2304 * 2);
  float*  out_mm     = (float*)  carve((size_t)NPIX * 256 * 4);

  // (1) gap + attention scale (parallel two-stage reduction)
  dfa_zero<<<4, 256, 0, stream>>>(gap, 4 * 256);
  dfa_gap_partial<<<dim3(4, 16), 256, 0, stream>>>(fine, gap);
  dfa_attn<<<4, 256, 0, stream>>>(gap, fs_attn_w, scale);

  // (2a) scaled fine -> bf16 A1
  dfa_prep_a1<<<NPIX, 256, 0, stream>>>(fine, scale, A1);

  // weight transposes to [N][K] bf16
  dfa_transpose_w<<<(256 * 256 + 255) / 256, 256, 0, stream>>>(fs_conv_w, Bt_fsconv, 256, 256, 256);
  dfa_transpose_w<<<(512 * 512 + 255) / 256, 256, 0, stream>>>(offset_w, Bt_off, 512, 512, 512);
  for (int tap = 0; tap < 9; ++tap) {
    dfa_transpose_w<<<(NPAD_OM * 512 + 255) / 256, 256, 0, stream>>>(
        om_w + (size_t)tap * 512 * 216, Bt_om + (size_t)tap * NPAD_OM * 512,
        512, 216, NPAD_OM);
  }
  dfa_transpose_w<<<(256 * 2304 + 255) / 256, 256, 0, stream>>>(dcn_kernel, Bt_dcn, 2304, 256, 256);

  // (3a) upsample coarse; also fills fc[:,256:]
  dfa_upsample<<<NPIX, 256, 0, stream>>>(coarse, cup, fc);

  // (2b) G1: fine_cal = A1 @ fs_conv_w     [16384,256] x [256,256]
  dfa_gemm<float, 2, 4><<<dim3(NPIX / 32, 256 / 64), 32, 0, stream>>>(
      A1, Bt_fsconv, fine_cal, NPIX, 256, 256);

  // fc[:,:256] = bf16(fine_cal)
  dfa_cvt_strided<<<NPIX, 256, 0, stream>>>(fine_cal, fc, 256, 512);

  // (3b) G2: align = fc @ offset_w -> bf16  [16384,512] x [512,512]
  dfa_gemm<__bf16, 2, 4><<<dim3(NPIX / 32, 512 / 64), 32, 0, stream>>>(
      fc, Bt_off, align_bf, NPIX, 512, 512);

  // (4) 3x3 conv -> om (padded N=224)
  dfa_conv3x3<<<dim3(NPIX / 32, NPAD_OM / 32), 32, 0, stream>>>(align_bf, Bt_om, om_b, om);

  // (5) deformable columns
  dfa_dcn_cols<<<dim3(NPIX, 9), dim3(32, 8), 0, stream>>>(om, cup, cols);

  // G4: out_mm = cols @ dcn_kernel   [16384,2304] x [2304,256]
  dfa_gemm<float, 2, 4><<<dim3(NPIX / 32, 256 / 64), 32, 0, stream>>>(
      cols, Bt_dcn, out_mm, NPIX, 256, 2304);

  // (6) relu + bias + residual
  dfa_epilogue<<<NPIX, 256, 0, stream>>>(out_mm, dcn_bias, fine_cal, out);
}
